// FusedOnlineAttention_30236569764232
// MI455X (gfx1250) — matrix operand order repaired and checked by reading, not compiled, not run on hardware
//
#include <hip/hip_runtime.h>
#include <hip/hip_bf16.h>

typedef __attribute__((ext_vector_type(16))) _Float16 v16h;
typedef __attribute__((ext_vector_type(8)))  _Float16 v8h;
typedef __attribute__((ext_vector_type(8)))  float    v8f;

#define BATCH 2
#define SQ    2048
#define SK    2048
#define NH    16
#define HD    64
// 1/sqrt(64) * log2(e): softmax computed in base-2 (v_exp_f32 is exp2)
#define QSCALE 0.1803368801111204f

// LDS row strides (in halves): 16B-aligned runs, rotating bank pattern.
#define KSTR 72
#define VSTR 40
#define PSTR 40

#define DPP8_SEL(a,b,c,d,e,f,g,h) \
    ((a)|((b)<<3)|((c)<<6)|((d)<<9)|((e)<<12)|((f)<<15)|((g)<<18)|((h)<<21))

template <int SEL>
__device__ __forceinline__ float dpp8f(float x) {
    return __builtin_bit_cast(float,
        __builtin_amdgcn_mov_dpp8(__builtin_bit_cast(int, x), SEL));
}
__device__ __forceinline__ float mirror16(float x) {
    // DPP16 ROW_MIRROR (0x140): lane i <-> 15-i within each 16-lane row
    return __builtin_bit_cast(float,
        __builtin_amdgcn_update_dpp(0, __builtin_bit_cast(int, x),
                                    0x140, 0xf, 0xf, true));
}
// max across each 16-lane half-group, pure VALU
__device__ __forceinline__ float rowmax16(float x) {
    x = fmaxf(x, dpp8f<DPP8_SEL(1,0,3,2,5,4,7,6)>(x));  // xor 1
    x = fmaxf(x, dpp8f<DPP8_SEL(2,3,0,1,6,7,4,5)>(x));  // xor 2
    x = fmaxf(x, dpp8f<DPP8_SEL(4,5,6,7,0,1,2,3)>(x));  // xor 4
    x = fmaxf(x, mirror16(x));                          // cross 8-groups
    return x;
}

__device__ __forceinline__ v16h make_v16(v8h lo, v8h hi) {
    v16h r;
#pragma unroll
    for (int i = 0; i < 8; ++i) { r[i] = lo[i]; r[i + 8] = hi[i]; }
    return r;
}

__global__ __launch_bounds__(128)
void flash_attn_fwd(const float* __restrict__ Q,
                    const float* __restrict__ K,
                    const float* __restrict__ V,
                    const int*   __restrict__ AMASK,
                    float* __restrict__ OUT) {
    // ping-pong K/V staging -> single barrier per iteration
    __shared__ _Float16 lds_k[2][32 * KSTR];     // K chunk, [key][d]
    __shared__ _Float16 lds_vT[2][HD * VSTR];    // V chunk transposed, [d][key]
    __shared__ _Float16 lds_p[4][16 * PSTR];     // per-wave P tile, [m][k]

    const int tid  = threadIdx.x;
    const int wave = tid >> 5;
    const int lane = tid & 31;
    const int n    = lane & 15;
    const int hi   = lane >> 4;

    const int q0 = blockIdx.x * 64;
    const int h  = blockIdx.y;
    const int b  = blockIdx.z;
    const int qbase_wave = q0 + wave * 16;

    // staging mapping for cooperative K/V loads
    const int key   = tid >> 2;          // 0..31
    const int dbase = (tid & 3) * 16;    // 0,16,32,48

    // ---- Q stripe -> registers, A layout, prescaled ----
    v16h qa[2];
    {
        const float* qp = Q + (((size_t)b * SQ + (qbase_wave + n)) * NH + h) * HD;
#pragma unroll
        for (int c = 0; c < 2; ++c) {
#pragma unroll
            for (int r = 0; r < 2; ++r) {
                const int d = 32 * c + 16 * r + 8 * hi;
                float4 f0 = *(const float4*)(qp + d);
                float4 f1 = *(const float4*)(qp + d + 4);
                qa[c][r * 8 + 0] = (_Float16)(f0.x * QSCALE);
                qa[c][r * 8 + 1] = (_Float16)(f0.y * QSCALE);
                qa[c][r * 8 + 2] = (_Float16)(f0.z * QSCALE);
                qa[c][r * 8 + 3] = (_Float16)(f0.w * QSCALE);
                qa[c][r * 8 + 4] = (_Float16)(f1.x * QSCALE);
                qa[c][r * 8 + 5] = (_Float16)(f1.y * QSCALE);
                qa[c][r * 8 + 6] = (_Float16)(f1.z * QSCALE);
                qa[c][r * 8 + 7] = (_Float16)(f1.w * QSCALE);
            }
        }
    }

    v16h ones;
#pragma unroll
    for (int i = 0; i < 16; ++i) ones[i] = (_Float16)1.0f;

    float mstate[8];
    v8f acc[4], acc_l;
#pragma unroll
    for (int j = 0; j < 8; ++j) { mstate[j] = -1e30f; acc_l[j] = 0.0f; }
#pragma unroll
    for (int o = 0; o < 4; ++o)
#pragma unroll
        for (int j = 0; j < 8; ++j) acc[o][j] = 0.0f;

    const int kend = q0 + 64;

    // ---- pipelined staging registers: prefetch chunk 0 ----
    float4 gk[4], gv[4];
    {
        const size_t row = ((size_t)b * SK + key) * NH + h;
        const float* kp = K + row * HD + dbase;
        const float* vp = V + row * HD + dbase;
#pragma unroll
        for (int i = 0; i < 4; ++i) { gk[i] = *(const float4*)(kp + 4 * i); }
#pragma unroll
        for (int i = 0; i < 4; ++i) { gv[i] = *(const float4*)(vp + 4 * i); }
    }

    int buf = 0;
    for (int kk = 0; kk < kend; kk += 32) {
        // ---- convert + store staged chunk into LDS buffer `buf` ----
#pragma unroll
        for (int i = 0; i < 4; ++i) {
            _Float16* dst = &lds_k[buf][key * KSTR + dbase + 4 * i];
            dst[0] = (_Float16)gk[i].x; dst[1] = (_Float16)gk[i].y;
            dst[2] = (_Float16)gk[i].z; dst[3] = (_Float16)gk[i].w;
        }
#pragma unroll
        for (int i = 0; i < 4; ++i) {
            const int d0 = dbase + 4 * i;
            lds_vT[buf][(d0 + 0) * VSTR + key] = (_Float16)gv[i].x;
            lds_vT[buf][(d0 + 1) * VSTR + key] = (_Float16)gv[i].y;
            lds_vT[buf][(d0 + 2) * VSTR + key] = (_Float16)gv[i].z;
            lds_vT[buf][(d0 + 3) * VSTR + key] = (_Float16)gv[i].w;
        }

        // ---- issue next chunk's global loads; they land under compute ----
        if (kk + 32 < kend) {
            const size_t row = ((size_t)b * SK + (kk + 32 + key)) * NH + h;
            const float* kp = K + row * HD + dbase;
            const float* vp = V + row * HD + dbase;
#pragma unroll
            for (int i = 0; i < 4; ++i) { gk[i] = *(const float4*)(kp + 4 * i); }
#pragma unroll
            for (int i = 0; i < 4; ++i) { gv[i] = *(const float4*)(vp + 4 * i); }
        }

        __syncthreads();   // buffer `buf` ready; also fences buf^1 reuse

        // ---- S = Qs . K^T : two 16x16 tiles ----
        v8f s[2];
#pragma unroll
        for (int t = 0; t < 2; ++t) {
            v8f c0;
#pragma unroll
            for (int j = 0; j < 8; ++j) c0[j] = 0.0f;
#pragma unroll
            for (int c = 0; c < 2; ++c) {
                const v8h* bp =
                    (const v8h*)&lds_k[buf][(16 * t + n) * KSTR + 32 * c + 8 * hi];
                v16h bb = make_v16(bp[0], bp[2]);
                c0 = __builtin_amdgcn_wmma_f32_16x16x32_f16(
                        false, qa[c], false, bb, (short)0, c0, false, false);
            }
            s[t] = c0;
        }

        // ---- mask + online softmax (base-2; row sums via WMMA) ----
        int am0 = AMASK[(size_t)b * SK + kk + n];
        int am1 = AMASK[(size_t)b * SK + kk + 16 + n];
#pragma unroll
        for (int j = 0; j < 8; ++j) {
            const int qrow = qbase_wave + 8 * hi + j;
            const int k0 = kk + n, k1 = kk + 16 + n;
            float s0 = ((k0 <= qrow) && (am0 == 0)) ? s[0][j] : -1e30f;
            float s1 = ((k1 <= qrow) && (am1 == 0)) ? s[1][j] : -1e30f;

            const float r  = rowmax16(fmaxf(s0, s1));
            const float nm = fmaxf(mstate[j], r);
            const float alpha = __builtin_amdgcn_exp2f(mstate[j] - nm);
            const float p0 = __builtin_amdgcn_exp2f(s0 - nm);
            const float p1 = __builtin_amdgcn_exp2f(s1 - nm);
            mstate[j] = nm;

            acc[0][j] *= alpha; acc[1][j] *= alpha;
            acc[2][j] *= alpha; acc[3][j] *= alpha;
            acc_l[j]  *= alpha;

            const int m = 8 * hi + j;
            lds_p[wave][m * PSTR + n]      = (_Float16)p0;
            lds_p[wave][m * PSTR + 16 + n] = (_Float16)p1;
        }

        // lds_p is per-wave; same-wave DS ops are in-order -> no barrier needed
        const v8h* pp0 = (const v8h*)&lds_p[wave][n * PSTR + 8 * hi];
        const v8h* pp1 = (const v8h*)&lds_p[wave][n * PSTR + 16 + 8 * hi];
        v16h pa = make_v16(pp0[0], pp1[0]);

        // ---- row sums: acc_l += P . ones ----
        acc_l = __builtin_amdgcn_wmma_f32_16x16x32_f16(
                    false, pa, false, ones, (short)0, acc_l, false, false);

        // ---- O += P . V ----
#pragma unroll
        for (int o = 0; o < 4; ++o) {
            const v8h* vp =
                (const v8h*)&lds_vT[buf][(16 * o + n) * VSTR + 8 * hi];
            v16h vb = make_v16(vp[0], vp[2]);
            acc[o] = __builtin_amdgcn_wmma_f32_16x16x32_f16(
                        false, pa, false, vb, (short)0, acc[o], false, false);
        }

        buf ^= 1;
    }

    // ---- epilogue: O / l ----
#pragma unroll
    for (int j = 0; j < 8; ++j) {
        const float inv = 1.0f / acc_l[j];
        const int qrow = qbase_wave + 8 * hi + j;
        float* op = OUT + (((size_t)b * SQ + qrow) * NH + h) * HD;
#pragma unroll
        for (int o = 0; o < 4; ++o)
            op[16 * o + n] = acc[o][j] * inv;
    }
}

extern "C" void kernel_launch(void* const* d_in, const int* in_sizes, int n_in,
                              void* d_out, int out_size, void* d_ws, size_t ws_size,
                              hipStream_t stream) {
    const float* Q = (const float*)d_in[0];
    const float* K = (const float*)d_in[1];
    const float* V = (const float*)d_in[2];
    const int*   M = (const int*)d_in[3];
    float* O = (float*)d_out;

    dim3 grid(SQ / 64, NH, BATCH);
    dim3 block(128);
    flash_attn_fwd<<<grid, block, 0, stream>>>(Q, K, V, M, O);
}
